// GaussProjection_71347996721426
// MI455X (gfx1250) — matrix operand order repaired and checked by reading, not compiled, not run on hardware
//
#include <hip/hip_runtime.h>

// ---------------------------------------------------------------------------
// GaussProjection on MI455X (gfx1250):
//   out[4096,256] = (x[4096,32000] @ F[32000,128]) @ weight.T[128,256]
// F[v,n] = cos(2*pi*(n+1)*v/V) (n<64)  |  -sin(2*pi*(n-63)*v/V) (n>=64)
// Memory-bound on reading x (524 MB -> ~22.5us @ 23.3 TB/s); f16 WMMA
// (16x16x32, f32 accum) keeps compute (~33.5 GFLOP) under the HBM floor.
// GEMM1 inner loop is software-pipelined with an UNCONDITIONAL steady state
// (branch-free prefetch) so the compiler emits partial s_wait_loadcnt and
// every wave keeps a full iteration of b128 loads in flight.
// ---------------------------------------------------------------------------

typedef __attribute__((ext_vector_type(16))) _Float16 v16h;
typedef __attribute__((ext_vector_type(8)))  _Float16 v8h;
typedef __attribute__((ext_vector_type(8)))  float    v8f;
typedef __attribute__((ext_vector_type(4)))  float    v4f;

namespace gp {
constexpr int B  = 2;
constexpr int S  = 2048;
constexpr int V  = 32000;
constexpr int M  = B * S;        // 4096 rows
constexpr int N1 = 128;          // 2*n_freq (GEMM1 N, GEMM2 K)
constexpr int C  = 256;          // channels (GEMM2 N)
constexpr int KSTEPS = V / 32;   // 1000 K-steps of 32
constexpr int SPLITS = 10;       // split-K factor for GEMM1 (KCH even)
constexpr int KCH    = KSTEPS / SPLITS;  // 100 K-steps per split
}

// ---------------------------------------------------------------------------
// Kernel 1: build F in WMMA B-fragment layout (f16).
// B fragment (32x16, f16), per ISA: lanes 0-15: K in {0..7,16..23}, N=lane;
// lanes 16-31: K in {8..15,24..31}, N=lane-16. Stored lane-major: fragment index
// (kb*8+nt), then lane*16 halves (h = K%8 + 8*(K/16)).
// ---------------------------------------------------------------------------
__global__ void gp_build_F(_Float16* __restrict__ F) {
  int idx = blockIdx.x * 256 + threadIdx.x;
  if (idx >= gp::V * gp::N1) return;
  int v = idx >> 7;          // vocab index
  int n = idx & 127;         // output component
  int k = (n & 63) + 1;      // harmonic 1..64
  int m = (k * v) % gp::V;   // exact integer phase (fits in int)
  float theta = (float)m * (6.28318530717958647692f / (float)gp::V);
  float s, c;
  __sincosf(theta, &s, &c);
  float val = (n < 64) ? c : -s;   // real: cos ; imag: -sin (e^{-i t})
  int kb = v >> 5, kr = v & 31;
  int nt = n >> 4, nr = n & 15;
  int lane = nr + (((kr >> 3) & 1) << 4);
  int h    = (kr & 7) + ((kr >> 4) << 3);
  F[(((size_t)kb * 8 + nt) * 32 + lane) * 16 + h] = (_Float16)val;
}

// ---------------------------------------------------------------------------
// GEMM1 helpers: clause-friendly loads + fused convert/MMA.
// A (16x32 f16) per ISA: lanes 0-15 hold row M=lane, K in {0..7,16..23};
// lanes 16-31 hold the same rows, K in {8..15,24..31}.
// ---------------------------------------------------------------------------
__device__ __forceinline__ void gp_loadA(const float* __restrict__ xrow,
                                         int ks, int hi, v4f a[4]) {
  const int kbase = ks * 32 + hi * 8;
  a[0] = *(const v4f*)(xrow + kbase);
  a[1] = *(const v4f*)(xrow + kbase + 4);
  a[2] = *(const v4f*)(xrow + kbase + 16);
  a[3] = *(const v4f*)(xrow + kbase + 20);
}

__device__ __forceinline__ void gp_loadB(const _Float16* __restrict__ F,
                                         int ks, int nhalf, int lane,
                                         v16h b[4]) {
  const v16h* Bbase = (const v16h*)F + (size_t)ks * 8 * 32;
#pragma unroll
  for (int t = 0; t < 4; ++t)
    b[t] = Bbase[(size_t)(nhalf * 4 + t) * 32 + lane];
}

__device__ __forceinline__ void gp_mma4(v8f acc[4], const v4f a[4],
                                        const v16h b[4]) {
  v16h A;
  A[0]  = (_Float16)a[0].x; A[1]  = (_Float16)a[0].y;
  A[2]  = (_Float16)a[0].z; A[3]  = (_Float16)a[0].w;
  A[4]  = (_Float16)a[1].x; A[5]  = (_Float16)a[1].y;
  A[6]  = (_Float16)a[1].z; A[7]  = (_Float16)a[1].w;
  A[8]  = (_Float16)a[2].x; A[9]  = (_Float16)a[2].y;
  A[10] = (_Float16)a[2].z; A[11] = (_Float16)a[2].w;
  A[12] = (_Float16)a[3].x; A[13] = (_Float16)a[3].y;
  A[14] = (_Float16)a[3].z; A[15] = (_Float16)a[3].w;
#pragma unroll
  for (int t = 0; t < 4; ++t)
    acc[t] = __builtin_amdgcn_wmma_f32_16x16x32_f16(
        false, A, false, b[t], (short)0, acc[t], false, false);
}

// ---------------------------------------------------------------------------
// Kernel 2: GEMM1, split-K, branch-free software pipeline (ping/pong).
// Wave tile: M=16 x N=64 (4 WMMA tiles). Block = 8 waves = 4 M-tiles x
// 2 N-halves. Grid = (64, SPLITS). KCH=100 (even) K-steps per wave.
// ---------------------------------------------------------------------------
__global__ void __launch_bounds__(256) gp_gemm1(
    const float* __restrict__ x, const _Float16* __restrict__ F,
    float* __restrict__ Ypart) {
  const int lane  = threadIdx.x & 31;
  const int wave  = threadIdx.x >> 5;
  const int mtile = blockIdx.x * 4 + (wave >> 1);   // 0..255
  const int nhalf = wave & 1;                       // 0..1
  const int split = blockIdx.y;                     // 0..SPLITS-1
  const int row0  = mtile * 16;
  const int l15   = lane & 15;
  const int hi    = lane >> 4;

  const float* xrow = x + (size_t)(row0 + l15) * gp::V;

  v8f acc[4] = {};

  const int ks0 = split * gp::KCH;

  v4f  aP[4], aQ[4];
  v16h bP[4], bQ[4];
  // Prologue: fill both pipeline stages.
  gp_loadA(xrow, ks0,     hi, aP); gp_loadB(F, ks0,     nhalf, lane, bP);
  gp_loadA(xrow, ks0 + 1, hi, aQ); gp_loadB(F, ks0 + 1, nhalf, lane, bQ);

  // Steady state: completely unconditional body (KCH is even, so i+3 <= KCH-1
  // whenever i+2 < KCH). This lets the compiler use partial loadcnt waits.
  for (int i = 0; i + 2 < gp::KCH; i += 2) {
    gp_mma4(acc, aP, bP);                           // consume step i
    gp_loadA(xrow, ks0 + i + 2, hi, aP);            // prefetch step i+2
    gp_loadB(F, ks0 + i + 2, nhalf, lane, bP);
    gp_mma4(acc, aQ, bQ);                           // consume step i+1
    gp_loadA(xrow, ks0 + i + 3, hi, aQ);            // prefetch step i+3
    gp_loadB(F, ks0 + i + 3, nhalf, lane, bQ);
  }
  // Drain: last two K-steps.
  gp_mma4(acc, aP, bP);
  gp_mma4(acc, aQ, bQ);

  // D layout (16x16 f32): VGPR r -> (M=r, N=lane) lanes 0-15; (M=r+8, N=lane-16).
  float* yp = Ypart + ((size_t)split * gp::M + row0) * gp::N1;
#pragma unroll
  for (int t = 0; t < 4; ++t) {
    const int col = (nhalf * 4 + t) * 16 + l15;
#pragma unroll
    for (int r = 0; r < 8; ++r) {
      yp[(size_t)(r + hi * 8) * gp::N1 + col] = acc[t][r];
    }
  }
}

// ---------------------------------------------------------------------------
// Kernel 3: reduce split-K slabs, convert to f16.
// ---------------------------------------------------------------------------
__global__ void gp_reduce_y(const float* __restrict__ Ypart,
                            _Float16* __restrict__ Yred) {
  int idx = blockIdx.x * 256 + threadIdx.x;
  if (idx >= gp::M * gp::N1) return;
  float s = 0.f;
#pragma unroll
  for (int p = 0; p < gp::SPLITS; ++p)
    s += Ypart[(size_t)p * gp::M * gp::N1 + idx];
  Yred[idx] = (_Float16)s;
}

// ---------------------------------------------------------------------------
// Kernel 4: GEMM2  out[4096,256] = Y[4096,128] @ weight.T.
// Wave tile M=16 x N=64, K = 4 steps of 32. B built on the fly from weight
// (row n of weight = column n of W^T), 16 contiguous f32 per lane, L2-hot.
// ---------------------------------------------------------------------------
__global__ void __launch_bounds__(256) gp_gemm2(
    const _Float16* __restrict__ Y, const float* __restrict__ W,
    float* __restrict__ out) {
  const int lane = threadIdx.x & 31;
  const int wave = threadIdx.x >> 5;
  const int task = blockIdx.x * 8 + wave;   // 0..1023
  const int mtile = task >> 2;              // 0..255
  const int nq    = task & 3;               // 0..3 (16 N-tiles / 4)
  const int row0  = mtile * 16;
  const int l15   = lane & 15;
  const int hi    = lane >> 4;

  const _Float16* yrow = Y + (size_t)(row0 + l15) * gp::N1;

  v8f acc[4] = {};

#pragma unroll
  for (int ks = 0; ks < 4; ++ks) {
    const int kbase = ks * 32 + hi * 16;
    v8h ylo = *(const v8h*)(yrow + kbase);        // K offsets +0..7
    v8h yhi = *(const v8h*)(yrow + kbase + 8);    // K offsets +8..15
    v16h A;
#pragma unroll
    for (int i = 0; i < 8; ++i) { A[i] = ylo[i]; A[8 + i] = yhi[i]; }

#pragma unroll
    for (int t = 0; t < 4; ++t) {
      const int n = (nq * 4 + t) * 16 + l15;
      const float* wr = W + (size_t)n * gp::N1 + ks * 32 + hi * 16;
      v4f w0 = *(const v4f*)wr;
      v4f w1 = *(const v4f*)(wr + 4);
      v4f w2 = *(const v4f*)(wr + 8);
      v4f w3 = *(const v4f*)(wr + 12);
      v16h Bf;
      Bf[0]  = (_Float16)w0.x; Bf[1]  = (_Float16)w0.y;
      Bf[2]  = (_Float16)w0.z; Bf[3]  = (_Float16)w0.w;
      Bf[4]  = (_Float16)w1.x; Bf[5]  = (_Float16)w1.y;
      Bf[6]  = (_Float16)w1.z; Bf[7]  = (_Float16)w1.w;
      Bf[8]  = (_Float16)w2.x; Bf[9]  = (_Float16)w2.y;
      Bf[10] = (_Float16)w2.z; Bf[11] = (_Float16)w2.w;
      Bf[12] = (_Float16)w3.x; Bf[13] = (_Float16)w3.y;
      Bf[14] = (_Float16)w3.z; Bf[15] = (_Float16)w3.w;
      acc[t] = __builtin_amdgcn_wmma_f32_16x16x32_f16(
          false, A, false, Bf, (short)0, acc[t], false, false);
    }
  }

#pragma unroll
  for (int t = 0; t < 4; ++t) {
    const int col = (nq * 4 + t) * 16 + l15;
#pragma unroll
    for (int r = 0; r < 8; ++r) {
      out[(size_t)(row0 + r + hi * 8) * gp::C + col] = acc[t][r];
    }
  }
}

// ---------------------------------------------------------------------------
// Launcher.  Workspace map (all regions fully overwritten every call):
//   [0,       8 MiB)   F fragments (f16, 8,192,000 B used)
//   [8 MiB,  28 MiB)   Ypart: SPLITS(10) x 4096 x 128 f32 (20,971,520 B)
//   [30 MiB, 31 MiB)   Yred:  4096 x 128 f16 (1,048,576 B)
// Total ~31 MiB of d_ws.
// ---------------------------------------------------------------------------
extern "C" void kernel_launch(void* const* d_in, const int* in_sizes, int n_in,
                              void* d_out, int out_size, void* d_ws, size_t ws_size,
                              hipStream_t stream) {
  (void)in_sizes; (void)n_in; (void)out_size; (void)ws_size;
  const float* x = (const float*)d_in[0];   // [2, 2048, 32000] f32
  const float* w = (const float*)d_in[1];   // [256, 128] f32
  float* out = (float*)d_out;               // [2, 2048, 256] f32

  char* ws = (char*)d_ws;
  _Float16* F     = (_Float16*)ws;
  float*    Ypart = (float*)(ws + (size_t)8 * 1024 * 1024);
  _Float16* Yred  = (_Float16*)(ws + (size_t)30 * 1024 * 1024);

  gp_build_F<<<(gp::V * gp::N1 + 255) / 256, 256, 0, stream>>>(F);
  dim3 g1(gp::M / 16 / 4, gp::SPLITS);            // (64, 10)
  gp_gemm1<<<g1, 256, 0, stream>>>(x, F, Ypart);
  gp_reduce_y<<<(gp::M * gp::N1 + 255) / 256, 256, 0, stream>>>(Ypart, Yred);
  gp_gemm2<<<(gp::M / 16) * 4 / 8, 256, 0, stream>>>(Yred, w, out);  // 128 blocks
}
